// FeatureEmbeddingNet_77867757076594
// MI455X (gfx1250) — matrix operand order repaired and checked by reading, not compiled, not run on hardware
//
#include <hip/hip_runtime.h>
#include <hip/hip_bf16.h>
#include <cstdint>
#include <cstddef>

#define KNN 20

typedef __attribute__((ext_vector_type(16))) __bf16 v16bf;
typedef __attribute__((ext_vector_type(8)))  float  v8f;

// ---------------------------------------------------------------------------
// WMMA fragment loaders (wave32; contiguous k order shared by A and B, not the table order of CDNA5 ISA §7.12.2)
// A: 16x32 bf16, row m striped in-lane. lane l: m=l&15, hi=l>>4,
//    K = [hi*16, hi*16+16)                          -> one contiguous run of 16
// B: 32x16 bf16, B[k][n] = src[n*stride + k]. lane l: n=l&15,
//    K = [(l>>4)*16, +16)                          -> one contiguous run of 16
// ---------------------------------------------------------------------------
static __device__ __forceinline__ v16bf load_a_frag(const __bf16* __restrict__ src,
                                                    int stride, int kb, int lane) {
  int m = lane & 15, hi = lane >> 4;
  const __bf16* p = src + (size_t)m * stride + kb;
  v16bf a;
#pragma unroll
  for (int i = 0; i < 16; ++i) {
    a[i] = p[hi * 16 + i];
  }
  return a;
}

static __device__ __forceinline__ v16bf load_b_frag(const __bf16* __restrict__ src,
                                                    int stride, int kb, int lane) {
  int n = lane & 15, koff = (lane >> 4) * 16;
  const __bf16* p = src + (size_t)n * stride + kb + koff;
  v16bf b;
#pragma unroll
  for (int i = 0; i < 16; ++i) b[i] = p[i];
  return b;
}

// order-preserving float <-> uint mapping for atomic max over mixed signs
static __device__ __forceinline__ unsigned fkey(float f) {
  unsigned u = __float_as_uint(f);
  return (u & 0x80000000u) ? ~u : (u | 0x80000000u);
}
static __device__ __forceinline__ float unfkey(unsigned u) {
  unsigned b = (u & 0x80000000u) ? (u & 0x7fffffffu) : ~u;
  return __uint_as_float(b);
}

// ---------------------------------------------------------------------------
// Fused Gram-matrix + top-k KNN. Block = 256 thr (8 waves) = one 16-row strip.
// Each wave owns one 16-col tile of the 128-col iteration; pd tile -> LDS;
// per-row sorted top-20 maintained in LDS. D = f_rows . f_cols^T via WMMA.
// Assumes N % 16 == 0 (N = 10000).
// ---------------------------------------------------------------------------
template<int CPAD>
__global__ __launch_bounds__(256) void knn_kernel(
    const __bf16* __restrict__ fbf, int stride,
    const float* __restrict__ norms, int N,
    int* __restrict__ idxout)
{
  constexpr int NKC = CPAD / 32;
  __shared__ float pdbuf[16][128];
  __shared__ float x2r[16];
  __shared__ float tkd[16][KNN];
  __shared__ int   tki[16][KNN];

  const int tid  = threadIdx.x;
  const int lane = tid & 31;
  const int wave = tid >> 5;
  const int rowBase = blockIdx.x * 16;

  if (tid < 16) {
    x2r[tid] = norms[rowBase + tid];
#pragma unroll
    for (int j = 0; j < KNN; ++j) { tkd[tid][j] = -3.0e38f; tki[tid][j] = 0; }
  }
  __syncthreads();

  v16bf afrag[NKC];
#pragma unroll
  for (int c = 0; c < NKC; ++c)
    afrag[c] = load_a_frag(fbf + (size_t)rowBase * stride, stride, c * 32, lane);

  const int n  = lane & 15;
  const int hi = lane >> 4;

  for (int colIter = 0; colIter < N; colIter += 128) {
    const int colBase = colIter + wave * 16;       // wave-uniform -> EXEC full
    const int ncols = (N - colIter < 128) ? (N - colIter) : 128;
    if (colBase < N) {
      v8f acc = {};
#pragma unroll
      for (int c = 0; c < NKC; ++c) {
        v16bf b = load_b_frag(fbf + (size_t)colBase * stride, stride, c * 32, lane);
        acc = __builtin_amdgcn_wmma_f32_16x16x32_bf16(
            false, afrag[c], false, b, (short)0, acc, false, false);
      }
      const float x2c = norms[colBase + n];
#pragma unroll
      for (int r = 0; r < 8; ++r) {
        int mm = r + 8 * hi;
        pdbuf[mm][wave * 16 + n] = 2.0f * acc[r] - x2r[mm] - x2c;
      }
    }
    __syncthreads();
    if (tid < 16) {                                 // per-row insertion top-k
      float* lst = tkd[tid];
      int*   li  = tki[tid];
      for (int j = 0; j < ncols; ++j) {
        float v = pdbuf[tid][j];
        if (v > lst[KNN - 1]) {
          int pos = KNN - 1;
          while (pos > 0 && lst[pos - 1] < v) {
            lst[pos] = lst[pos - 1]; li[pos] = li[pos - 1]; --pos;
          }
          lst[pos] = v; li[pos] = colIter + j;
        }
      }
    }
    __syncthreads();
  }
  if (tid < 16) {
#pragma unroll
    for (int j = 0; j < KNN; ++j)
      idxout[(size_t)(rowBase + tid) * KNN + j] = tki[tid][j];
  }
}

// ---------------------------------------------------------------------------
// EdgeConv GEMM. Block = 256 thr = 4 points x 20 neighbors = 80 edge rows.
// A tile [80 x Kpad] built in LDS (bf16) from gathered [nb-ctr, ctr].
// B = wT [Cout][Kpad] bf16 from global (L2-resident).
// mode 0: accumulate per-channel sum/sumsq of h for BN (global atomics).
// mode 1: y = elu(h*scale+shift); max over k via LDS atomic-max keys; write
//         xc (f32) + xcbf (bf16) columns and next-layer squared norms.
// ---------------------------------------------------------------------------
__global__ __launch_bounds__(256) void edge_conv_kernel(
    const float* __restrict__ fin, int finStride,
    const int* __restrict__ idx,
    const __bf16* __restrict__ wT,
    int Cin, int Kpad, int Cout, int mode,
    float* __restrict__ gsum, float* __restrict__ gsq,
    const float* __restrict__ scale, const float* __restrict__ shift,
    float* __restrict__ xc, __bf16* __restrict__ xcbf,
    float* __restrict__ normsOut, int outOfs, int N)
{
  __shared__ __bf16   ebuf[80 * 256];
  __shared__ float    chansum[256];
  __shared__ float    chansq[256];
  __shared__ unsigned obuf[4 * 256];
  __shared__ float    nrm[4];

  const int tid  = threadIdx.x;
  const int lane = tid & 31;
  const int wave = tid >> 5;
  const int pointBase = blockIdx.x * 4;

  if (mode == 0) {
    chansum[tid] = 0.f; chansq[tid] = 0.f;
  } else {
    for (int e = tid; e < 4 * 256; e += 256) obuf[e] = 0u;
    if (tid < 4) nrm[tid] = 0.f;
  }

  // build A tile: e[row] = [f[nb]-f[n] (Cin), f[n] (Cin), zero pad]
  const int tot_e = 80 * Kpad;
  for (int e = tid; e < tot_e; e += 256) {
    int row = e / Kpad, k = e - row * Kpad;
    int p = row / KNN, kk = row - p * KNN;
    int nPt = pointBase + p;
    float v = 0.f;
    if (k < Cin) {
      int nb = idx[(size_t)nPt * KNN + kk];
      v = fin[(size_t)nb * finStride + k] - fin[(size_t)nPt * finStride + k];
    } else if (k < 2 * Cin) {
      v = fin[(size_t)nPt * finStride + (k - Cin)];
    }
    ebuf[e] = (__bf16)v;
  }
  __syncthreads();

  const int nTiles = Cout >> 4;
  const int totT = 5 * nTiles;        // 5 M-tiles (80 rows) x nTiles
  const int nn = lane & 15;
  const int hi = lane >> 4;

  for (int t = wave; t < totT; t += 8) {
    int mT = t / nTiles, nT = t - mT * nTiles;
    v8f acc = {};
    for (int kb = 0; kb < Kpad; kb += 32) {
      v16bf a = load_a_frag(ebuf + (size_t)mT * 16 * Kpad, Kpad, kb, lane);
      v16bf b = load_b_frag(wT + (size_t)nT * 16 * Kpad, Kpad, kb, lane);
      acc = __builtin_amdgcn_wmma_f32_16x16x32_bf16(
          false, a, false, b, (short)0, acc, false, false);
    }
    int ch = nT * 16 + nn;
    if (mode == 0) {
      float s = 0.f, sq = 0.f;
#pragma unroll
      for (int r = 0; r < 8; ++r) { float y = acc[r]; s += y; sq += y * y; }
      atomicAdd(&chansum[ch], s);
      atomicAdd(&chansq[ch], sq);
    } else {
      float sc = scale[ch], sh = shift[ch];
#pragma unroll
      for (int r = 0; r < 8; ++r) {
        int mm = mT * 16 + r + 8 * hi;   // edge row 0..79
        int p  = mm / KNN;
        float y = acc[r] * sc + sh;
        y = (y > 0.f) ? y : (expf(y) - 1.f);
        atomicMax(&obuf[p * Cout + ch], fkey(y));
      }
    }
  }
  __syncthreads();

  if (mode == 0) {
    for (int c = tid; c < Cout; c += 256) {
      atomicAdd(&gsum[c], chansum[c]);
      atomicAdd(&gsq[c],  chansq[c]);
    }
  } else {
    for (int e = tid; e < 4 * Cout; e += 256) {
      int p = e / Cout, ch = e - p * Cout;
      float v = unfkey(obuf[e]);
      size_t o = (size_t)(pointBase + p) * 512 + outOfs + ch;
      xc[o]   = v;
      xcbf[o] = (__bf16)v;
      atomicAdd(&nrm[p], v * v);
    }
    __syncthreads();
    if (tid < 4) normsOut[pointBase + tid] = nrm[tid];
  }
}

// ---------------------------------------------------------------------------
// FC: h5 = xcbf[N,512] @ fcT^T + bias, plus per-channel BN stats.
// Block = 256 thr = 16 rows x 256 cols (16 N-tiles over 8 waves), K = 512.
// ---------------------------------------------------------------------------
__global__ __launch_bounds__(256) void fc_gemm_kernel(
    const __bf16* __restrict__ xcbf, const __bf16* __restrict__ fcT,
    const float* __restrict__ fc_b, float* __restrict__ h5,
    float* __restrict__ gsum, float* __restrict__ gsq, int N)
{
  __shared__ float chansum[256];
  __shared__ float chansq[256];
  const int tid  = threadIdx.x;
  const int lane = tid & 31;
  const int wave = tid >> 5;
  const int rowBase = blockIdx.x * 16;

  chansum[tid] = 0.f; chansq[tid] = 0.f;
  __syncthreads();

  const int nn = lane & 15, hi = lane >> 4;
  for (int nT = wave; nT < 16; nT += 8) {
    v8f acc = {};
    for (int kb = 0; kb < 512; kb += 32) {
      v16bf a = load_a_frag(xcbf + (size_t)rowBase * 512, 512, kb, lane);
      v16bf b = load_b_frag(fcT + (size_t)nT * 16 * 512, 512, kb, lane);
      acc = __builtin_amdgcn_wmma_f32_16x16x32_bf16(
          false, a, false, b, (short)0, acc, false, false);
    }
    int ch = nT * 16 + nn;
    float bias = fc_b[ch];
    float s = 0.f, sq = 0.f;
#pragma unroll
    for (int r = 0; r < 8; ++r) {
      float y = acc[r] + bias;
      h5[(size_t)(rowBase + r + 8 * hi) * 256 + ch] = y;
      s += y; sq += y * y;
    }
    atomicAdd(&chansum[ch], s);
    atomicAdd(&chansq[ch],  sq);
  }
  __syncthreads();
  atomicAdd(&gsum[tid], chansum[tid]);
  atomicAdd(&gsq[tid],  chansq[tid]);
}

// ---------------------------------------------------------------------------
// small helper kernels
// ---------------------------------------------------------------------------
__global__ void bn_finalize_kernel(const float* __restrict__ gsum,
                                   const float* __restrict__ gsq,
                                   const float* __restrict__ g,
                                   const float* __restrict__ b,
                                   float invCount,
                                   float* __restrict__ scale,
                                   float* __restrict__ shift, int C)
{
  int c = blockIdx.x * blockDim.x + threadIdx.x;
  if (c < C) {
    float m   = gsum[c] * invCount;
    float var = gsq[c] * invCount - m * m;
    float sc  = g[c] * rsqrtf(var + 1e-5f);
    scale[c] = sc;
    shift[c] = b[c] - m * sc;
  }
}

__global__ void final_elu_kernel(const float* __restrict__ h5,
                                 const float* __restrict__ scale,
                                 const float* __restrict__ shift,
                                 float* __restrict__ out, int total)
{
  int i = blockIdx.x * blockDim.x + threadIdx.x;
  if (i < total) {
    int ch = i & 255;
    float y = h5[i] * scale[ch] + shift[ch];
    out[i] = (y > 0.f) ? y : (expf(y) - 1.f);
  }
}

__global__ void prep_input_kernel(const float* __restrict__ x,
                                  __bf16* __restrict__ f0bf,
                                  float* __restrict__ norms0, int N)
{
  int n = blockIdx.x * blockDim.x + threadIdx.x;
  if (n < N) {
    float s = 0.f;
#pragma unroll
    for (int c = 0; c < 6; ++c) {
      float v = x[(size_t)n * 6 + c];
      s += v * v;
      f0bf[(size_t)n * 32 + c] = (__bf16)v;
    }
#pragma unroll
    for (int c = 6; c < 32; ++c) f0bf[(size_t)n * 32 + c] = (__bf16)0.f;
    norms0[n] = s;
  }
}

// transpose + bf16 + zero-pad K: wT[ch][k] = w[k][ch] (k < K), else 0
__global__ void prep_wT_kernel(const float* __restrict__ w,
                               __bf16* __restrict__ wT,
                               int K, int Kpad, int Cout)
{
  int e = blockIdx.x * blockDim.x + threadIdx.x;
  if (e < Cout * Kpad) {
    int ch = e / Kpad, k = e - ch * Kpad;
    wT[e] = (k < K) ? (__bf16)w[(size_t)k * Cout + ch] : (__bf16)0.f;
  }
}

__global__ void zero_kernel(float* __restrict__ buf, int n) {
  int i = blockIdx.x * blockDim.x + threadIdx.x;
  if (i < n) buf[i] = 0.f;
}

// ---------------------------------------------------------------------------
// host launch
// ---------------------------------------------------------------------------
extern "C" void kernel_launch(void* const* d_in, const int* in_sizes, int n_in,
                              void* d_out, int out_size, void* d_ws, size_t ws_size,
                              hipStream_t stream)
{
  const float* x    = (const float*)d_in[0];
  const float* w1   = (const float*)d_in[1];
  const float* g1   = (const float*)d_in[2];
  const float* b1   = (const float*)d_in[3];
  const float* w2   = (const float*)d_in[4];
  const float* g2   = (const float*)d_in[5];
  const float* b2   = (const float*)d_in[6];
  const float* w3   = (const float*)d_in[7];
  const float* g3   = (const float*)d_in[8];
  const float* b3   = (const float*)d_in[9];
  const float* w4   = (const float*)d_in[10];
  const float* g4   = (const float*)d_in[11];
  const float* b4   = (const float*)d_in[12];
  const float* fc_w = (const float*)d_in[13];
  const float* fc_b = (const float*)d_in[14];
  const float* g5   = (const float*)d_in[15];
  const float* b5   = (const float*)d_in[16];
  // d_in[17] = k (== 20, baked into KNN)

  const int N = in_sizes[0] / 6;   // 10000; code assumes N % 16 == 0

  // workspace carve (~43 MB total)
  char* p = (char*)d_ws;
  auto carve = [&](size_t bytes) -> char* {
    char* r = p; p += (bytes + 255) & ~(size_t)255; return r;
  };
  float*  xc     = (float*) carve((size_t)N * 512 * sizeof(float));   // concat feats
  __bf16* xcbf   = (__bf16*)carve((size_t)N * 512 * sizeof(__bf16));  // bf16 mirror
  __bf16* f0bf   = (__bf16*)carve((size_t)N * 32  * sizeof(__bf16));
  float*  h5     = (float*) carve((size_t)N * 256 * sizeof(float));
  int*    idx    = (int*)   carve((size_t)N * KNN * sizeof(int));
  float*  norms0 = (float*) carve((size_t)N * sizeof(float));
  float*  norms1 = (float*) carve((size_t)N * sizeof(float));
  float*  norms2 = (float*) carve((size_t)N * sizeof(float));
  float*  norms3 = (float*) carve((size_t)N * sizeof(float));
  float*  norms4 = (float*) carve((size_t)N * sizeof(float));
  __bf16* wT1    = (__bf16*)carve((size_t)64  * 32  * sizeof(__bf16));
  __bf16* wT2    = (__bf16*)carve((size_t)64  * 128 * sizeof(__bf16));
  __bf16* wT3    = (__bf16*)carve((size_t)128 * 128 * sizeof(__bf16));
  __bf16* wT4    = (__bf16*)carve((size_t)256 * 256 * sizeof(__bf16));
  __bf16* fcT    = (__bf16*)carve((size_t)256 * 512 * sizeof(__bf16));
  float*  stats  = (float*) carve(1536 * sizeof(float));   // gsum/gsq per BN
  float*  ss     = (float*) carve(1536 * sizeof(float));   // scale/shift per BN

  float *gsum1 = stats + 0,    *gsq1 = stats + 64;
  float *gsum2 = stats + 128,  *gsq2 = stats + 192;
  float *gsum3 = stats + 256,  *gsq3 = stats + 384;
  float *gsum4 = stats + 512,  *gsq4 = stats + 768;
  float *gsum5 = stats + 1024, *gsq5 = stats + 1280;
  float *sc1 = ss + 0,    *sh1 = ss + 64;
  float *sc2 = ss + 128,  *sh2 = ss + 192;
  float *sc3 = ss + 256,  *sh3 = ss + 384;
  float *sc4 = ss + 512,  *sh4 = ss + 768;
  float *sc5 = ss + 1024, *sh5 = ss + 1280;

  dim3 blk(256);
  const int rowBlocks = N / 16;
  const int ptBlocks  = N / 4;
  const float invNK = 1.f / ((float)N * (float)KNN);

  prep_input_kernel<<<(N + 255) / 256, blk, 0, stream>>>(x, f0bf, norms0, N);
  prep_wT_kernel<<<(64 * 32  + 255) / 256, blk, 0, stream>>>(w1,  wT1, 12,  32,  64);
  prep_wT_kernel<<<(64 * 128 + 255) / 256, blk, 0, stream>>>(w2,  wT2, 128, 128, 64);
  prep_wT_kernel<<<(128 * 128 + 255) / 256, blk, 0, stream>>>(w3, wT3, 128, 128, 128);
  prep_wT_kernel<<<(256 * 256 + 255) / 256, blk, 0, stream>>>(w4, wT4, 256, 256, 256);
  prep_wT_kernel<<<(256 * 512 + 255) / 256, blk, 0, stream>>>(fc_w, fcT, 512, 512, 256);
  zero_kernel<<<(1536 + 255) / 256, blk, 0, stream>>>(stats, 1536);

  // ---- Layer 1: Cin=6 (pad 32), Cout=64, out cols [0,64)
  knn_kernel<32><<<rowBlocks, blk, 0, stream>>>(f0bf, 32, norms0, N, idx);
  edge_conv_kernel<<<ptBlocks, blk, 0, stream>>>(x, 6, idx, wT1, 6, 32, 64, 0,
      gsum1, gsq1, nullptr, nullptr, nullptr, nullptr, nullptr, 0, N);
  bn_finalize_kernel<<<1, 64, 0, stream>>>(gsum1, gsq1, g1, b1, invNK, sc1, sh1, 64);
  edge_conv_kernel<<<ptBlocks, blk, 0, stream>>>(x, 6, idx, wT1, 6, 32, 64, 1,
      nullptr, nullptr, sc1, sh1, xc, xcbf, norms1, 0, N);

  // ---- Layer 2: Cin=64, Cout=64, in cols [0,64) -> out cols [64,128)
  knn_kernel<64><<<rowBlocks, blk, 0, stream>>>(xcbf + 0, 512, norms1, N, idx);
  edge_conv_kernel<<<ptBlocks, blk, 0, stream>>>(xc + 0, 512, idx, wT2, 64, 128, 64, 0,
      gsum2, gsq2, nullptr, nullptr, nullptr, nullptr, nullptr, 0, N);
  bn_finalize_kernel<<<1, 64, 0, stream>>>(gsum2, gsq2, g2, b2, invNK, sc2, sh2, 64);
  edge_conv_kernel<<<ptBlocks, blk, 0, stream>>>(xc + 0, 512, idx, wT2, 64, 128, 64, 1,
      nullptr, nullptr, sc2, sh2, xc, xcbf, norms2, 64, N);

  // ---- Layer 3: Cin=64, Cout=128, in cols [64,128) -> out cols [128,256)
  knn_kernel<64><<<rowBlocks, blk, 0, stream>>>(xcbf + 64, 512, norms2, N, idx);
  edge_conv_kernel<<<ptBlocks, blk, 0, stream>>>(xc + 64, 512, idx, wT3, 64, 128, 128, 0,
      gsum3, gsq3, nullptr, nullptr, nullptr, nullptr, nullptr, 0, N);
  bn_finalize_kernel<<<2, 64, 0, stream>>>(gsum3, gsq3, g3, b3, invNK, sc3, sh3, 128);
  edge_conv_kernel<<<ptBlocks, blk, 0, stream>>>(xc + 64, 512, idx, wT3, 64, 128, 128, 1,
      nullptr, nullptr, sc3, sh3, xc, xcbf, norms3, 128, N);

  // ---- Layer 4: Cin=128, Cout=256, in cols [128,256) -> out cols [256,512)
  knn_kernel<128><<<rowBlocks, blk, 0, stream>>>(xcbf + 128, 512, norms3, N, idx);
  edge_conv_kernel<<<ptBlocks, blk, 0, stream>>>(xc + 128, 512, idx, wT4, 128, 256, 256, 0,
      gsum4, gsq4, nullptr, nullptr, nullptr, nullptr, nullptr, 0, N);
  bn_finalize_kernel<<<4, 64, 0, stream>>>(gsum4, gsq4, g4, b4, invNK, sc4, sh4, 256);
  edge_conv_kernel<<<ptBlocks, blk, 0, stream>>>(xc + 128, 512, idx, wT4, 128, 256, 256, 1,
      nullptr, nullptr, sc4, sh4, xc, xcbf, norms4, 256, N);

  // ---- FC + BN(axis 0) + ELU
  fc_gemm_kernel<<<rowBlocks, blk, 0, stream>>>(xcbf, fcT, fc_b, h5, gsum5, gsq5, N);
  bn_finalize_kernel<<<4, 64, 0, stream>>>(gsum5, gsq5, g5, b5, 1.f / (float)N, sc5, sh5, 256);
  final_elu_kernel<<<(N * 256 + 255) / 256, blk, 0, stream>>>(h5, sc5, sh5, (float*)d_out, N * 256);

  (void)n_in; (void)out_size; (void)ws_size;
}
